// CNN2Trans_21895743275478
// MI455X (gfx1250) — matrix operand order repaired and checked by reading, not compiled, not run on hardware
//
#include <hip/hip_runtime.h>
#include <hip/hip_bf16.h>
#include <math.h>

typedef __bf16 bf16_t;
typedef __attribute__((ext_vector_type(16))) __bf16 v16bf;
typedef __attribute__((ext_vector_type(8)))  float  v8f;
typedef __attribute__((ext_vector_type(4)))  unsigned int v4u;
typedef __attribute__((ext_vector_type(8)))  int v8i;
typedef __attribute__((ext_vector_type(4)))  int v4i;

union FragU { uint4 q[2]; v16bf v; };

// ---------------------------------------------------------------------------
// Tensor Data Mover: 2D bf16 tile load global -> LDS with row padding.
// D# built per CDNA5 ISA 8.3/8.4. pad_interval p: interval = (2<<p) dwords,
// pad_amount_m1 a: pad = (a+1) dwords appended per interval.
// ---------------------------------------------------------------------------
__device__ __forceinline__ void tdm_load_2d(void* lds_dst, const void* gsrc,
                                            unsigned tile_cols, unsigned tile_rows,
                                            unsigned row_stride_elems,
                                            unsigned pad_interval, unsigned pad_amount_m1) {
    unsigned long long ga = (unsigned long long)gsrc;
    v4u g0;
    g0[0] = 1u;                                            // count=1, user mode
    g0[1] = (unsigned)(unsigned long long)lds_dst;         // lds_addr (bytes)
    g0[2] = (unsigned)ga;                                  // global_addr[31:0]
    g0[3] = (unsigned)((ga >> 32) & 0x1FFFFFFull) | (2u << 30);  // addr[56:32] | type=2
    unsigned flags = (1u << 16)                            // data_size = 2 bytes
                   | (1u << 20)                            // pad_enable
                   | (pad_interval << 22) | (pad_amount_m1 << 25);
    v8i g1;
    g1[0] = (int)flags;
    g1[1] = (int)((row_stride_elems & 0xFFFFu) << 16);     // tensor_dim0[15:0]
    g1[2] = (int)(((row_stride_elems >> 16) & 0xFFFFu) | ((tile_rows & 0xFFFFu) << 16));
    g1[3] = (int)(((tile_rows >> 16) & 0xFFFFu) | ((tile_cols & 0xFFFFu) << 16));
    g1[4] = (int)(tile_rows & 0xFFFFu);                    // tile_dim1 (tile_dim2=0)
    g1[5] = (int)row_stride_elems;                         // tensor_dim0_stride[31:0]
    g1[6] = 0;
    g1[7] = 0;
    v4i z4 = {};
    v8i z8 = {};
    __builtin_amdgcn_tensor_load_to_lds(g0, g1, z4, z4, z8, 0);
}

// A fragment (16x32 bf16, ISA 7.12.2): lane m=l&15, half=l>>4.
__device__ __forceinline__ v16bf load_a_frag(const bf16_t* sA, int rowBase, int k0, int stride) {
    const int lane = threadIdx.x & 31;
    const int m = lane & 15, half = lane >> 4;
    const bf16_t* p = sA + (rowBase + m) * stride + k0 + half * 8;
    FragU u;
    u.q[0] = *reinterpret_cast<const uint4*>(p);
    u.q[1] = *reinterpret_cast<const uint4*>(p + 16);
    return u.v;
}

// B fragment (32x16 bf16) from W^T image in LDS (row n holds K contiguous).
__device__ __forceinline__ v16bf load_b_frag(const bf16_t* sWT, int nBase, int k0, int stride) {
    const int lane = threadIdx.x & 31;
    const int n = nBase + (lane & 15), half = lane >> 4;
    const bf16_t* p = sWT + n * stride + k0 + half * 8;
    FragU u;
    u.q[0] = *reinterpret_cast<const uint4*>(p);
    u.q[1] = *reinterpret_cast<const uint4*>(p + 16);
    return u.v;
}

__device__ __forceinline__ v8f wmma_bf16(v16bf a, v16bf b, v8f c) {
    return __builtin_amdgcn_wmma_f32_16x16x32_bf16(false, a, false, b, (short)0, c, false, false);
}

// ---------------------------------------------------------------------------
// Prep kernels (one-time bf16 packing so GEMM staging is pure TDM DMA)
// ---------------------------------------------------------------------------
__global__ __launch_bounds__(256) void cvt_bf16_kernel(const float4* __restrict__ src,
                                                       uint2* __restrict__ dst, int n4) {
    int i = blockIdx.x * 256 + threadIdx.x;
    if (i < n4) {
        float4 f = src[i];
        union { bf16_t h[4]; uint2 u; } o;
        o.h[0] = (bf16_t)f.x; o.h[1] = (bf16_t)f.y;
        o.h[2] = (bf16_t)f.z; o.h[3] = (bf16_t)f.w;
        dst[i] = o.u;
    }
}

// W[K x N] f32 -> W^T[N x K] bf16
__global__ __launch_bounds__(256) void wtrans_kernel(const float* __restrict__ src,
                                                     bf16_t* __restrict__ dst, int K, int N) {
    int i = blockIdx.x * 256 + threadIdx.x;
    if (i < K * N) {
        int k = i / N, n = i % N;
        dst[(size_t)n * K + k] = (bf16_t)src[i];
    }
}

// x_cnn[b][c][2h+m][2w+n] f32 -> KV[(loc*4 + m*2+n)][c] bf16 (contiguous rows)
__global__ __launch_bounds__(256) void kv_gather_kernel(const float* __restrict__ xc,
                                                        bf16_t* __restrict__ kv) {
    int i = blockIdx.x * 256 + threadIdx.x;      // 8*128*128*128 elements
    int wwp = i & 127, hhp = (i >> 7) & 127, bc = i >> 14;
    int c = bc & 127, b = bc >> 7;
    int loc = (b << 12) + ((hhp >> 1) << 6) + (wwp >> 1);
    int nn = ((hhp & 1) << 1) | (wwp & 1);
    kv[(((size_t)loc * 4 + nn) << 7) + c] = (bf16_t)xc[i];
}

// ---------------------------------------------------------------------------
// TDM double-buffered GEMM: out[M x N] = act(A[M x K](bf16) @ W^T + bias (+res))
// EPI: 0=bias | 1=bias+residual | 2=bias+gelu | 3=bias,f32 store
// ---------------------------------------------------------------------------
template <int N, int K, int EPI>
__global__ __launch_bounds__(256) void gemm_tdm_kernel(
    const bf16_t* __restrict__ A, const bf16_t* __restrict__ WT,
    const float* __restrict__ bias, const float* __restrict__ resid,
    bf16_t* __restrict__ outB, float* __restrict__ outF) {
    constexpr int KT = K / 32, NT = N / 16;
    __shared__ __align__(16) bf16_t sA[2][128 * 40];
    __shared__ __align__(16) bf16_t sW[2][N * 40];
    const int rowBase = blockIdx.x * 128;
    const int t = threadIdx.x;
    const int wv = t >> 5, lane = t & 31;

    auto issue = [&](int kt, int b) {
        // 64B rows -> pad_interval 3 (16 dw), pad 4 dw -> stride 40 elems
        tdm_load_2d(&sA[b][0], A + (size_t)rowBase * K + kt * 32, 32, 128, K, 3, 3);
        tdm_load_2d(&sW[b][0], WT + kt * 32, 32, N, K, 3, 3);
    };
    if (wv == 0) {
        issue(0, 0);
        if (KT > 1) issue(1, 1);
    }

    v8f acc[NT] = {};
    for (int kt = 0; kt < KT; ++kt) {
        if (wv == 0) {
            if (kt + 1 < KT) __builtin_amdgcn_s_wait_tensorcnt(2);  // slab kt done
            else             __builtin_amdgcn_s_wait_tensorcnt(0);
        }
        __syncthreads();
        v16bf af = load_a_frag(&sA[kt & 1][0], wv * 16, 0, 40);
#pragma unroll
        for (int nt = 0; nt < NT; ++nt) {
            v16bf bfg = load_b_frag(&sW[kt & 1][0], nt * 16, 0, 40);
            acc[nt] = wmma_bf16(af, bfg, acc[nt]);
        }
        __syncthreads();
        if (wv == 0 && kt + 2 < KT) issue(kt + 2, kt & 1);
    }

    // epilogue (C/D: VGPR r -> M = r + 8*(lane>>4), N = lane&15)
#pragma unroll
    for (int nt = 0; nt < NT; ++nt) {
        const int col = nt * 16 + (lane & 15);
        const float bcol = bias[col];
#pragma unroll
        for (int r = 0; r < 8; ++r) {
            const int row = rowBase + wv * 16 + r + (lane >> 4) * 8;
            float x = acc[nt][r] + bcol;
            if constexpr (EPI == 1) x += resid[(size_t)row * N + col];
            if constexpr (EPI == 2) x = 0.5f * x * (1.0f + erff(x * 0.70710678118654752f));
            if constexpr (EPI == 3) outF[(size_t)row * N + col] = x;
            else                    outB[(size_t)row * N + col] = (bf16_t)x;
        }
    }
}

// ---------------------------------------------------------------------------
// Fused window attention, TDM-staged. WG = 64 locations (256 KV rows).
// ---------------------------------------------------------------------------
__global__ __launch_bounds__(256) void win_attn_kernel(
    const bf16_t* __restrict__ kvbuf, const bf16_t* __restrict__ qbuf,
    const bf16_t* __restrict__ WkT, const float* __restrict__ bk,
    const bf16_t* __restrict__ WvT, const float* __restrict__ bv,
    bf16_t* __restrict__ obuf) {
    __shared__ __align__(16) bf16_t sKV[256 * 136];
    __shared__ __align__(16) bf16_t sK[256 * 40];
    __shared__ __align__(16) bf16_t sV[256 * 40];
    __shared__ __align__(16) bf16_t sWk[2][32 * 136];
    __shared__ __align__(16) bf16_t sWv[2][32 * 136];
    const int locBase = blockIdx.x * 64;
    const int t = threadIdx.x;
    const int wv = t >> 5, lane = t & 31;

    auto issueHead = [&](int g, int b) {
        // 256B rows -> pad_interval 5 (64 dw), pad 4 dw -> stride 136 elems
        tdm_load_2d(&sWk[b][0], WkT + (size_t)g * 32 * 128, 128, 32, 128, 5, 3);
        tdm_load_2d(&sWv[b][0], WvT + (size_t)g * 32 * 128, 128, 32, 128, 5, 3);
    };
    if (wv == 0) {
        tdm_load_2d(&sKV[0], kvbuf + (size_t)locBase * 4 * 128, 128, 256, 128, 5, 3);
        issueHead(0, 0);
        issueHead(1, 1);
    }

    for (int g = 0; g < 8; ++g) {
        if (wv == 0) {
            if (g + 1 < 8) __builtin_amdgcn_s_wait_tensorcnt(2);  // KV + head g done
            else           __builtin_amdgcn_s_wait_tensorcnt(0);
        }
        __syncthreads();

        // K_g, V_g: 256x128 @ 128x32 -> 16 row-tiles x 2 n-tiles; 2 row-tiles/wave
        v8f aK[2][2] = {};
        v8f aV[2][2] = {};
        const bf16_t* wkp = &sWk[g & 1][0];
        const bf16_t* wvp = &sWv[g & 1][0];
#pragma unroll
        for (int kt = 0; kt < 4; ++kt) {
#pragma unroll
            for (int rt = 0; rt < 2; ++rt) {
                v16bf af = load_a_frag(&sKV[0], (wv * 2 + rt) * 16, kt * 32, 136);
#pragma unroll
                for (int nt = 0; nt < 2; ++nt) {
                    v16bf bkf = load_b_frag(wkp, nt * 16, kt * 32, 136);
                    aK[rt][nt] = wmma_bf16(af, bkf, aK[rt][nt]);
                    v16bf bvf = load_b_frag(wvp, nt * 16, kt * 32, 136);
                    aV[rt][nt] = wmma_bf16(af, bvf, aV[rt][nt]);
                }
            }
        }
#pragma unroll
        for (int rt = 0; rt < 2; ++rt) {
#pragma unroll
            for (int nt = 0; nt < 2; ++nt) {
                const int col = nt * 16 + (lane & 15);
                const float bkc = bk[g * 32 + col];
                const float bvc = bv[g * 32 + col];
#pragma unroll
                for (int r = 0; r < 8; ++r) {
                    const int row = (wv * 2 + rt) * 16 + r + (lane >> 4) * 8;
                    sK[row * 40 + col] = (bf16_t)(aK[rt][nt][r] + bkc);
                    sV[row * 40 + col] = (bf16_t)(aV[rt][nt][r] + bvc);
                }
            }
        }
        __syncthreads();
        // overlap next-next head DMA with the attention math below
        if (wv == 0 && g + 2 < 8) issueHead(g + 2, g & 1);

        // softmax_one attention: one thread per location (n=4, d=32)
        if (t < 64) {
            const int loc = locBase + t;
            const bf16_t* qp = qbuf + (size_t)loc * 256 + g * 32;
            float l0 = 0.f, l1 = 0.f, l2 = 0.f, l3 = 0.f;
#pragma unroll 8
            for (int d = 0; d < 32; ++d) {
                float qd = (float)qp[d];
                l0 += qd * (float)sK[(t * 4 + 0) * 40 + d];
                l1 += qd * (float)sK[(t * 4 + 1) * 40 + d];
                l2 += qd * (float)sK[(t * 4 + 2) * 40 + d];
                l3 += qd * (float)sK[(t * 4 + 3) * 40 + d];
            }
            const float scale = 0.17677669529663689f;  // 1/sqrt(32)
            l0 *= scale; l1 *= scale; l2 *= scale; l3 *= scale;
            float m = fmaxf(fmaxf(l0, l1), fmaxf(l2, l3));
            float e0 = __expf(l0 - m), e1 = __expf(l1 - m);
            float e2 = __expf(l2 - m), e3 = __expf(l3 - m);
            float inv = 1.0f / (1.0f + e0 + e1 + e2 + e3);  // softmax_one
            e0 *= inv; e1 *= inv; e2 *= inv; e3 *= inv;
            bf16_t* op = obuf + (size_t)loc * 256 + g * 32;
#pragma unroll 8
            for (int d = 0; d < 32; ++d) {
                float o = e0 * (float)sV[(t * 4 + 0) * 40 + d]
                        + e1 * (float)sV[(t * 4 + 1) * 40 + d]
                        + e2 * (float)sV[(t * 4 + 2) * 40 + d]
                        + e3 * (float)sV[(t * 4 + 3) * 40 + d];
                op[d] = (bf16_t)o;
            }
        }
    }
}

extern "C" void kernel_launch(void* const* d_in, const int* in_sizes, int n_in,
                              void* d_out, int out_size, void* d_ws, size_t ws_size,
                              hipStream_t stream) {
    const float* x_trans = (const float*)d_in[0];
    const float* x_cnn   = (const float*)d_in[1];
    const float* Wq  = (const float*)d_in[2];
    const float* bq  = (const float*)d_in[3];
    const float* Wk  = (const float*)d_in[4];
    const float* bk  = (const float*)d_in[5];
    const float* Wv  = (const float*)d_in[6];
    const float* bv  = (const float*)d_in[7];
    const float* Wp  = (const float*)d_in[8];
    const float* bp  = (const float*)d_in[9];
    const float* Wa1 = (const float*)d_in[10];
    const float* ba1 = (const float*)d_in[11];
    const float* Wa2 = (const float*)d_in[12];
    const float* ba2 = (const float*)d_in[13];

    char* ws = (char*)d_ws;
    const size_t MB = 1ull << 20;
    bf16_t* xtb   = (bf16_t*)(ws);              // 32768x256 bf16, 16 MB
    bf16_t* qbuf  = (bf16_t*)(ws + 16 * MB);    // 16 MB
    bf16_t* kvbuf = (bf16_t*)(ws + 32 * MB);    // 131072x128 bf16, 32 MB
    bf16_t* obuf  = (bf16_t*)(ws + 64 * MB);    // 16 MB
    bf16_t* ybuf  = (bf16_t*)(ws + 16 * MB);    // reuse q region (q dead after attn)
    bf16_t* hbuf  = (bf16_t*)(ws);              // reuse xtb region (dead after GEMM-q)
    bf16_t* WqT   = (bf16_t*)(ws + 80 * MB);
    bf16_t* WkT   = (bf16_t*)(ws + 80 * MB + 256 * 1024);
    bf16_t* WvT   = (bf16_t*)(ws + 80 * MB + 512 * 1024);
    bf16_t* WpT   = (bf16_t*)(ws + 80 * MB + 768 * 1024);
    bf16_t* Wa1T  = (bf16_t*)(ws + 80 * MB + 1024 * 1024);
    bf16_t* Wa2T  = (bf16_t*)(ws + 80 * MB + 1280 * 1024);

    dim3 blk(256);
    // ---- prep: bf16 packing / transposes / window gather ----
    cvt_bf16_kernel<<<8192, blk, 0, stream>>>((const float4*)x_trans, (uint2*)xtb, 2097152);
    wtrans_kernel<<<256, blk, 0, stream>>>(Wq, WqT, 256, 256);
    wtrans_kernel<<<128, blk, 0, stream>>>(Wk, WkT, 128, 256);
    wtrans_kernel<<<128, blk, 0, stream>>>(Wv, WvT, 128, 256);
    wtrans_kernel<<<256, blk, 0, stream>>>(Wp, WpT, 256, 256);
    wtrans_kernel<<<64,  blk, 0, stream>>>(Wa1, Wa1T, 256, 64);
    wtrans_kernel<<<64,  blk, 0, stream>>>(Wa2, Wa2T, 64, 256);
    kv_gather_kernel<<<65536, blk, 0, stream>>>(x_cnn, kvbuf);

    // ---- main pipeline ----
    // q = x_trans @ Wq + bq
    gemm_tdm_kernel<256, 256, 0><<<256, blk, 0, stream>>>(xtb, WqT, bq, nullptr, qbuf, nullptr);
    // fused window attention
    win_attn_kernel<<<512, blk, 0, stream>>>(kvbuf, qbuf, WkT, bk, WvT, bv, obuf);
    // y = attn_out @ Wp + bp + x_trans
    gemm_tdm_kernel<256, 256, 1><<<256, blk, 0, stream>>>(obuf, WpT, bp, x_trans, ybuf, nullptr);
    // h = gelu(y @ Wa1 + ba1)
    gemm_tdm_kernel<64, 256, 2><<<256, blk, 0, stream>>>(ybuf, Wa1T, ba1, nullptr, hbuf, nullptr);
    // xs = h @ Wa2 + ba2 -> f32 d_out
    gemm_tdm_kernel<256, 64, 3><<<256, blk, 0, stream>>>(hbuf, Wa2T, ba2, nullptr, nullptr, (float*)d_out);
}
